// MAESelfAttention_80728205296094
// MI455X (gfx1250) — matrix operand order repaired and checked by reading, not compiled, not run on hardware
//
#include <hip/hip_runtime.h>
#include <hip/hip_bf16.h>
#include <stdint.h>

// Problem constants (from reference)
#define Hh   12
#define Dh   64
#define Ss   2048
#define Bb   2
#define HID  768            // Hh*Dh
#define SP   2080           // padded key length: s+1=2049 -> multiple of 32
#define MROWS 4112          // b*(s+1)=4098 padded to multiple of 16
#define NCOLS 1536          // 2*HID
#define KDIM 768
#define NEGBIG (-10000.0f)

typedef __bf16 bf16x16 __attribute__((ext_vector_type(16)));
typedef float  f32x8   __attribute__((ext_vector_type(8)));

union ABreg { bf16x16 v; unsigned int u[8]; };

__device__ __forceinline__ unsigned short f2bf(float f) {
  unsigned int u = __float_as_uint(f);
  u += 0x7FFFu + ((u >> 16) & 1u);   // round-to-nearest-even
  return (unsigned short)(u >> 16);
}

// ---- pack x = concat(embx, hidden) -> bf16, rows padded/zeroed ----
__global__ void pack_x_kernel(const float* __restrict__ hidden,
                              const float* __restrict__ embx,
                              unsigned short* __restrict__ xb) {
  int i = blockIdx.x * blockDim.x + threadIdx.x;
  if (i >= MROWS * KDIM) return;
  int row = i / KDIM, col = i % KDIM;
  float v = 0.0f;
  if (row < Bb * (Ss + 1)) {
    int b = row / (Ss + 1), seq = row % (Ss + 1);
    v = (seq == 0) ? embx[(size_t)b * KDIM + col]
                   : hidden[((size_t)b * Ss + (seq - 1)) * KDIM + col];
  }
  xb[i] = f2bf(v);
}

// ---- pack W transposed: Wt[c][j] = bf16(W[j][c]) ----
__global__ void pack_wt_kernel(const float* __restrict__ W,
                               unsigned short* __restrict__ Wt) {
  int i = blockIdx.x * blockDim.x + threadIdx.x;
  if (i >= NCOLS * KDIM) return;
  int c = i / KDIM, j = i % KDIM;
  Wt[i] = f2bf(W[(size_t)j * NCOLS + c]);
}

// ---- pack Q (expanded_embx) -> bf16, same (b,s,hid) layout ----
__global__ void pack_q_kernel(const float* __restrict__ q,
                              unsigned short* __restrict__ Qb) {
  int i = blockIdx.x * blockDim.x + threadIdx.x;
  if (i >= Bb * Ss * HID) return;
  Qb[i] = f2bf(q[i]);
}

// ---- key visibility mask + zero K/Vt key-padding ----
__global__ void prep_aux_kernel(const unsigned char* __restrict__ am,
                                const unsigned char* __restrict__ mlm,
                                unsigned char* __restrict__ keymask,
                                unsigned short* __restrict__ Kbf,
                                unsigned short* __restrict__ Vt) {
  int i = blockIdx.x * blockDim.x + threadIdx.x;
  const int PADK = SP - (Ss + 1);            // 31
  const int NM = Bb * SP;                    // mask entries
  const int NK = Bb * Hh * PADK * Dh;        // K pad entries
  if (i < NM) {
    int b = i / SP, k = i % SP;
    unsigned char m;
    if (k == 0) m = 1;
    else if (k <= Ss) m = (am[(size_t)b * Ss + k - 1] && !mlm[(size_t)b * Ss + k - 1]) ? 1 : 0;
    else m = 0;
    keymask[i] = m;
  } else if (i < NM + NK) {
    int t = i - NM;
    int d = t % Dh; t /= Dh;
    int kp = t % PADK; t /= PADK;
    int bh = t;
    Kbf[((size_t)bh * SP + (Ss + 1) + kp) * Dh + d] = 0;
  } else if (i < NM + 2 * NK) {
    int t = i - NM - NK;
    int kp = t % PADK; t /= PADK;
    int dh = t;                               // bh*64+d, 0..1535
    Vt[(size_t)dh * SP + (Ss + 1) + kp] = 0;
  }
}

// ---- KV GEMM: one wave per 16x16 tile, bf16 WMMA, fp32 accumulate ----
__global__ void kv_gemm_kernel(const unsigned short* __restrict__ xb,
                               const unsigned short* __restrict__ Wt,
                               const float* __restrict__ bias,
                               unsigned short* __restrict__ Kbf,
                               unsigned short* __restrict__ Vt) {
  int wave = (int)((blockIdx.x * blockDim.x + threadIdx.x) >> 5);
  int lane = threadIdx.x & 31;
  int lo = lane & 15, hi = lane >> 4;
  const int NT = NCOLS / 16;                  // 96 column tiles
  int mt = wave / NT, nt = wave % NT;
  if (mt >= MROWS / 16) return;

  const unsigned short* arow = xb + (size_t)(mt * 16 + lo) * KDIM;  // A: row = lane%16
  const unsigned short* brow = Wt + (size_t)(nt * 16 + lo) * KDIM;  // B: col = lane%16

  f32x8 acc = {};
  for (int k0 = 0; k0 < KDIM; k0 += 32) {
    ABreg a, b;
#pragma unroll
    for (int v = 0; v < 8; ++v) {
      int ka = k0 + hi * 16 + v * 2;   // A 16x32 layout
      a.u[v] = *(const unsigned int*)(arow + ka);
      int kb = k0 + hi * 16 + v * 2;                        // B 32x16 layout
      b.u[v] = *(const unsigned int*)(brow + kb);
    }
    acc = __builtin_amdgcn_wmma_f32_16x16x32_bf16(false, a.v, false, b.v,
                                                  (short)0, acc, false, false);
  }

  int n = nt * 16 + lo;                        // output column 0..1535
  float bn = bias[n];
#pragma unroll
  for (int r = 0; r < 8; ++r) {
    int m = mt * 16 + hi * 8 + r;
    if (m >= Bb * (Ss + 1)) continue;
    int bidx = m / (Ss + 1), seq = m % (Ss + 1);
    unsigned short val = f2bf(acc[r] + bn);
    if (n < HID) {                             // K: (bh, key, d)
      int h = n / Dh, d = n % Dh;
      Kbf[(((size_t)(bidx * Hh + h)) * SP + seq) * Dh + d] = val;
    } else {                                   // V transposed: (bh, d, key)
      int c = n - HID, h = c / Dh, d = c % Dh;
      Vt[((size_t)(bidx * Hh + h) * Dh + d) * SP + seq] = val;
    }
  }
}

// ---- flash attention: one wave per 16 query rows ----
__global__ void attn_kernel(const unsigned short* __restrict__ Qb,
                            const unsigned short* __restrict__ Kbf,
                            const unsigned short* __restrict__ Vt,
                            const unsigned char* __restrict__ keymask,
                            float* __restrict__ out) {
  __shared__ unsigned short plds[4][16][32];   // per-wave P tile (C->A relayout)
  int w = threadIdx.x >> 5;
  int lane = threadIdx.x & 31;
  int lo = lane & 15, hi = lane >> 4;
  int gw = blockIdx.x * 4 + w;
  int qt = gw & 127;                           // 128 q-tiles of 16
  int bh = gw >> 7;                            // 0..23
  int b = bh / Hh, h = bh % Hh;

  // Q A-tiles (16x32 each over d=0..31 and d=32..63), resident for whole loop
  const unsigned short* qrow = Qb + ((size_t)(b * Ss + qt * 16 + lo)) * HID + h * Dh;
  ABreg a0, a1;
#pragma unroll
  for (int v = 0; v < 8; ++v) {
    int k = hi * 16 + v * 2;
    a0.u[v] = *(const unsigned int*)(qrow + k);
    a1.u[v] = *(const unsigned int*)(qrow + 32 + k);
  }

  const unsigned short* Kbase = Kbf + (size_t)bh * SP * Dh;
  const unsigned short* Vbase = Vt + (size_t)bh * Dh * SP;
  const unsigned char* km = keymask + (size_t)b * SP;

  float rmax[8], rsum[8];
  f32x8 O[4] = {};
#pragma unroll
  for (int r = 0; r < 8; ++r) { rmax[r] = -1e30f; rsum[r] = 0.0f; }
  const float scale = 0.125f;                  // D^-0.5

  for (int kt2 = 0; kt2 < SP / 32; ++kt2) {
    float s[2][8];
#pragma unroll
    for (int sub = 0; sub < 2; ++sub) {
      int kb = kt2 * 32 + sub * 16;
      const unsigned short* krow = Kbase + (size_t)(kb + lo) * Dh;  // B col = key
      ABreg b0, b1;
#pragma unroll
      for (int v = 0; v < 8; ++v) {
        int d = hi * 16 + v * 2;
        b0.u[v] = *(const unsigned int*)(krow + d);
        b1.u[v] = *(const unsigned int*)(krow + 32 + d);
      }
      f32x8 c = {};
      c = __builtin_amdgcn_wmma_f32_16x16x32_bf16(false, a0.v, false, b0.v,
                                                  (short)0, c, false, false);
      c = __builtin_amdgcn_wmma_f32_16x16x32_bf16(false, a1.v, false, b1.v,
                                                  (short)0, c, false, false);
#pragma unroll
      for (int r = 0; r < 8; ++r) {
        int key = kb + lo;
        int q = qt * 16 + hi * 8 + r;
        bool vis = km[key] && (key != q + 1);
        s[sub][r] = c[r] * scale + (vis ? 0.0f : NEGBIG);
      }
    }
    if (kt2 + 1 < SP / 32)                     // global_prefetch_b8 next K tile
      __builtin_prefetch(Kbase + (size_t)((kt2 + 1) * 32 + lane) * Dh, 0, 1);

    // online softmax (row m lives in half-wave hi==m/8, vgpr r==m%8)
    float alpha[8];
#pragma unroll
    for (int r = 0; r < 8; ++r) {
      float t = fmaxf(s[0][r], s[1][r]);
      t = fmaxf(t, __shfl_xor(t, 1));
      t = fmaxf(t, __shfl_xor(t, 2));
      t = fmaxf(t, __shfl_xor(t, 4));
      t = fmaxf(t, __shfl_xor(t, 8));
      float nm = fmaxf(rmax[r], t);
      alpha[r] = __expf(rmax[r] - nm);
      rmax[r] = nm;
      float p0 = __expf(s[0][r] - nm);
      float p1 = __expf(s[1][r] - nm);
      s[0][r] = p0; s[1][r] = p1;
      float ps = p0 + p1;
      ps += __shfl_xor(ps, 1);
      ps += __shfl_xor(ps, 2);
      ps += __shfl_xor(ps, 4);
      ps += __shfl_xor(ps, 8);
      rsum[r] = rsum[r] * alpha[r] + ps;
    }
#pragma unroll
    for (int dc = 0; dc < 4; ++dc)
#pragma unroll
      for (int r = 0; r < 8; ++r) O[dc][r] *= alpha[r];

    // P: C-layout -> A-layout via per-wave LDS tile
#pragma unroll
    for (int r = 0; r < 8; ++r) {
      int m = hi * 8 + r;
      plds[w][m][lo]      = f2bf(s[0][r]);
      plds[w][m][16 + lo] = f2bf(s[1][r]);
    }
    asm volatile("s_wait_dscnt 0x0" ::: "memory");
    ABreg pa;
#pragma unroll
    for (int v = 0; v < 8; ++v) {
      int k = hi * 16 + v * 2;
      pa.u[v] = *(const unsigned int*)(&plds[w][lo][k]);
    }
    // PV: contraction over 32 keys, N = 16-wide d chunks
#pragma unroll
    for (int dc = 0; dc < 4; ++dc) {
      const unsigned short* vrow = Vbase + (size_t)(dc * 16 + lo) * SP + kt2 * 32;
      ABreg bv;
#pragma unroll
      for (int v = 0; v < 8; ++v)
        bv.u[v] = *(const unsigned int*)(vrow + hi * 16 + v * 2);
      O[dc] = __builtin_amdgcn_wmma_f32_16x16x32_bf16(false, pa.v, false, bv.v,
                                                      (short)0, O[dc], false, false);
    }
  }

#pragma unroll
  for (int r = 0; r < 8; ++r) {
    int q = qt * 16 + hi * 8 + r;
    float inv = 1.0f / rsum[r];
    float* orow = out + ((size_t)(b * Ss + q)) * HID + h * Dh;
    orow[lo]      = O[0][r] * inv;
    orow[16 + lo] = O[1][r] * inv;
    orow[32 + lo] = O[2][r] * inv;
    orow[48 + lo] = O[3][r] * inv;
  }
}

extern "C" void kernel_launch(void* const* d_in, const int* in_sizes, int n_in,
                              void* d_out, int out_size, void* d_ws, size_t ws_size,
                              hipStream_t stream) {
  const float* hidden   = (const float*)d_in[0];
  const float* embx     = (const float*)d_in[1];
  const float* exp_embx = (const float*)d_in[2];
  const float* Wkv_w    = (const float*)d_in[3];
  const float* Wkv_b    = (const float*)d_in[4];
  const unsigned char* am  = (const unsigned char*)d_in[5];  // jnp bool = 1 byte
  const unsigned char* mlm = (const unsigned char*)d_in[6];
  float* out = (float*)d_out;

  char* ws = (char*)d_ws;
  size_t off = 0;
  auto alloc = [&](size_t bytes) {
    void* p = ws + off;
    off += (bytes + 255) & ~(size_t)255;
    return p;
  };
  unsigned short* xb  = (unsigned short*)alloc((size_t)MROWS * KDIM * 2);
  unsigned short* Wt  = (unsigned short*)alloc((size_t)NCOLS * KDIM * 2);
  unsigned short* Qb  = (unsigned short*)alloc((size_t)Bb * Ss * HID * 2);
  unsigned short* Kbf = (unsigned short*)alloc((size_t)Bb * Hh * SP * Dh * 2);
  unsigned short* Vt  = (unsigned short*)alloc((size_t)Bb * Hh * Dh * SP * 2);
  unsigned char* keymask = (unsigned char*)alloc((size_t)Bb * SP);
  (void)in_sizes; (void)n_in; (void)out_size; (void)ws_size;

  pack_x_kernel<<<(MROWS * KDIM + 255) / 256, 256, 0, stream>>>(hidden, embx, xb);
  pack_wt_kernel<<<(NCOLS * KDIM + 255) / 256, 256, 0, stream>>>(Wkv_w, Wt);
  pack_q_kernel<<<(Bb * Ss * HID + 255) / 256, 256, 0, stream>>>(exp_embx, Qb);
  int naux = Bb * SP + 2 * (Bb * Hh * (SP - (Ss + 1)) * Dh);
  prep_aux_kernel<<<(naux + 255) / 256, 256, 0, stream>>>(am, mlm, keymask, Kbf, Vt);
  int ntiles = (MROWS / 16) * (NCOLS / 16);           // 257 * 96
  kv_gemm_kernel<<<(ntiles + 7) / 8, 256, 0, stream>>>(xb, Wt, Wkv_b, Kbf, Vt);
  int nwaves = Bb * Hh * (Ss / 16);                   // 3072 waves
  attn_kernel<<<nwaves / 4, 128, 0, stream>>>(Qb, Kbf, Vt, keymask, out);
}
